// Model_352187318272
// MI455X (gfx1250) — matrix operand order repaired and checked by reading, not compiled, not run on hardware
//
#include <hip/hip_runtime.h>
#include <hip/hip_bf16.h>
#include <cstdint>
#include <cstddef>

typedef _Float16 f16;
typedef _Float16 v16h __attribute__((ext_vector_type(16)));
typedef float    v8f  __attribute__((ext_vector_type(8)));

#define EPSV 1e-4f

// ---------------------------------------------------------------- utilities
__global__ void k_zero(float* __restrict__ p, int n) {
  int i = blockIdx.x * blockDim.x + threadIdx.x;
  for (; i < n; i += gridDim.x * blockDim.x) p[i] = 0.f;
}

// ---------------------------------------------------------------- input conv (Cin = 1)
__global__ void k_conv_in(const float* __restrict__ x, const int* __restrict__ nb,
                          const float* __restrict__ w, float* __restrict__ out, int N) {
  int i = blockIdx.x * blockDim.x + threadIdx.x;
  int total = N * 32;
  for (; i < total; i += gridDim.x * blockDim.x) {
    int n = i >> 5, d = i & 31;
    float s = 0.f;
    #pragma unroll
    for (int k = 0; k < 27; ++k) {
      int idx = nb[n * 27 + k];
      if (idx < N) s += x[idx] * w[k * 32 + d];
    }
    out[i] = s;
  }
}

// ---------------------------------------------------------------- BN statistics (sum, sumsq)
__global__ void k_bn_stats(const float* __restrict__ x, float* __restrict__ stats,
                           int N, int C, int rowsPerBlock) {
  int r0 = blockIdx.x * rowsPerBlock;
  int r1 = r0 + rowsPerBlock; if (r1 > N) r1 = N;
  for (int c = threadIdx.x; c < C; c += blockDim.x) {
    float s = 0.f, s2 = 0.f;
    for (int r = r0; r < r1; ++r) {
      float v = x[(size_t)r * C + c];
      s += v; s2 += v * v;
    }
    atomicAdd(&stats[c], s);
    atomicAdd(&stats[C + c], s2);
  }
}

// ---------------------------------------------------------------- BN+ReLU, cast to f16, write zero sentinel row N
__global__ void k_bn_relu_cast(const float* __restrict__ x, const float* __restrict__ stats,
                               const float* __restrict__ g, const float* __restrict__ b,
                               f16* __restrict__ h, int N, int C) {
  size_t total = (size_t)(N + 1) * C;
  size_t stride = (size_t)gridDim.x * blockDim.x;
  float invN = 1.f / (float)N;
  for (size_t i = (size_t)blockIdx.x * blockDim.x + threadIdx.x; i < total; i += stride) {
    int c = (int)(i % (size_t)C);
    int r = (int)(i / (size_t)C);
    float y = 0.f;
    if (r < N) {
      float mu  = stats[c] * invN;
      float var = stats[C + c] * invN - mu * mu;
      float k   = g[c] * rsqrtf(var + EPSV);
      y = (x[i] - mu) * k + b[c];
      y = y > 0.f ? y : 0.f;
    }
    h[i] = (f16)y;
  }
}

// ---------------------------------------------------------------- pack f32 weights [Krows, Ncols] into per-lane
// WMMA B fragments (zero-padded to NT*16 cols):
// Bp[((kt*NT + nt)*32 + lane)*8 + j] = pack2(f16 of rows kt*32+16h+2j, +1 at col nt*16+(lane&15))
__global__ void k_pack_w(const float* __restrict__ w, unsigned* __restrict__ Bp,
                         int Krows, int Ncols, int NT) {
  int KT = Krows >> 5;
  int total = KT * NT * 32;
  for (int t = blockIdx.x * blockDim.x + threadIdx.x; t < total; t += gridDim.x * blockDim.x) {
    int lane = t & 31;
    int grp  = t >> 5;
    int nt   = grp % NT;
    int kt   = grp / NT;
    int n  = lane & 15, hh = lane >> 4;
    int col = nt * 16 + n;
    bool cv = col < Ncols;
    unsigned* dst = Bp + (size_t)t * 8;
    #pragma unroll
    for (int j = 0; j < 8; ++j) {
      int row = kt * 32 + 16 * hh + 2 * j;
      union { f16 h[2]; unsigned u; } pk;
      pk.h[0] = cv ? (f16)w[(size_t)row * Ncols + col] : (f16)0.f;
      pk.h[1] = cv ? (f16)w[(size_t)(row + 1) * Ncols + col] : (f16)0.f;
      dst[j] = pk.u;
    }
  }
}

// ---------------------------------------------------------------- gather + WMMA GEMM core
// h16: [(Nsrc+1), CIN] f16 (row Nsrc = zeros); nb: [M, 27]; Bp: packed B fragments; out: [M, COUT] f32
// Block = 128 threads (4 waves); one 16-row output tile per block; wave w handles col tiles w, w+4, w+8.
// A-tile staging uses global_load_async_to_lds_b64 (ASYNCcnt) with double-buffered LDS.
#define ATILE_STRIDE 20   // dwords per row (16 data + 4 pad): keeps b64/b128 alignment, spreads banks

template <int CIN, int COUT>
__global__ __launch_bounds__(128)
void k_gather_wmma(const f16* __restrict__ h16, const int* __restrict__ nb,
                   const unsigned* __restrict__ Bp, float* __restrict__ out,
                   int M, int Nsrc) {
  constexpr int KT   = (27 * CIN) / 32;   // K-steps of 32
  constexpr int NT   = COUT / 16;         // column tiles
  constexpr int NACC = (NT + 3) / 4;      // col tiles per wave
  constexpr int CPB  = CIN / 32;          // cin blocks per neighbor tap

  __shared__ __align__(16) unsigned Atile[2][16 * ATILE_STRIDE];

  const int t    = threadIdx.x;
  const int wave = __builtin_amdgcn_readfirstlane(t >> 5);  // SGPR: scalar guards around WMMA
  const int lane = t & 31;
  const int m    = lane & 15;
  const int hh   = lane >> 4;
  const int tile = blockIdx.x;

  const v8f vz = {0.f, 0.f, 0.f, 0.f, 0.f, 0.f, 0.f, 0.f};
  v8f acc[NACC];
  #pragma unroll
  for (int a = 0; a < NACC; ++a) acc[a] = vz;

  // staging assignment: 16 rows x 8 b64 chunks, one async b64 per thread per K-step
  const int rstage = t >> 3;          // 0..15
  const int dpair  = t & 7;           // 0..7  -> dwords 2*dpair, 2*dpair+1
  const int rg     = tile * 16 + rstage;
  const bool rvalid = rg < M;
  const int* nbrow = nb + (size_t)(rvalid ? rg : 0) * 27;   // clamped: load unconditionally

  // LDS byte addresses for this thread's staging slot in each buffer
  unsigned ldsA[2];
  ldsA[0] = (unsigned)(uintptr_t)(void*)&Atile[0][rstage * ATILE_STRIDE + dpair * 2];
  ldsA[1] = (unsigned)(uintptr_t)(void*)&Atile[1][rstage * ATILE_STRIDE + dpair * 2];

  auto stage = [&](int kt) {
    const int k  = kt / CPB;                       // compile-time divisor
    const int cb = kt - k * CPB;
    int idx = nbrow[k];
    if (!rvalid) idx = Nsrc;                       // sentinel row -> zeros (cndmask, no saveexec)
    const f16* gp = h16 + (unsigned)idx * CIN + cb * 32u + dpair * 4u;
    asm volatile("global_load_async_to_lds_b64 %0, %1, off"
                 :: "v"(ldsA[kt & 1]), "v"((unsigned long long)(uintptr_t)gp)
                 : "memory");
  };

  stage(0);
  for (int kt = 0; kt < KT; ++kt) {
    asm volatile("s_wait_asynccnt 0x0" ::: "memory");   // buffer kt&1 filled (per-wave)
    __syncthreads();                                    // visible to all waves
    if (kt + 1 < KT) {
      stage(kt + 1);                                    // fill other buffer asynchronously
      __builtin_prefetch(Bp + (((size_t)(kt + 1) * NT + wave) * 32 + lane) * 8, 0, 1);
    }

    const unsigned* cur = &Atile[kt & 1][0];
    // A fragment, contiguous k order: lane(m,h) -> row m dwords [8h .. 8h+7] (k = 16h..16h+15)
    union { v16h h; uint4 q[2]; } af;
    af.q[0] = ((const uint4*)cur)[m * (ATILE_STRIDE / 4) + 2 * hh];
    af.q[1] = ((const uint4*)cur)[m * (ATILE_STRIDE / 4) + 2 * hh + 1];

    #pragma unroll
    for (int a = 0; a < NACC; ++a) {
      const int nt = wave + 4 * a;
      if (nt < NT) {                                    // SGPR compare -> s_cbranch, EXEC untouched
        const uint4* bp = (const uint4*)(Bp + (((size_t)kt * NT + nt) * 32 + lane) * 8);
        union { v16h h; uint4 q[2]; } bf_;
        bf_.q[0] = bp[0];
        bf_.q[1] = bp[1];
        acc[a] = __builtin_amdgcn_wmma_f32_16x16x32_f16(
            /*neg_a=*/false, af.h, /*neg_b=*/false, bf_.h,
            /*c_mod=*/(short)0, acc[a], /*reuse_a=*/false, /*reuse_b=*/false);
      }
    }
    // no trailing barrier: each wave's dscnt wait (before its WMMA) precedes its
    // barrier signal, so the next overwrite of this buffer can't race in-flight reads.
  }

  // store: lane(m,h), vgpr v -> row = 16*tile + v + 8h, col = nt*16 + m
  const bool full = (tile * 16 + 16) <= M;              // block-uniform fast path
  #pragma unroll
  for (int a = 0; a < NACC; ++a) {
    const int nt = wave + 4 * a;
    if (nt < NT) {
      float* op = out + (size_t)(tile * 16 + 8 * hh) * COUT + nt * 16 + m;
      if (full) {
        #pragma unroll
        for (int vv = 0; vv < 8; ++vv) op[(size_t)vv * COUT] = acc[a][vv];
      } else {
        #pragma unroll
        for (int vv = 0; vv < 8; ++vv) {
          int row = tile * 16 + vv + 8 * hh;
          if (row < M) out[(size_t)row * COUT + nt * 16 + m] = acc[a][vv];
        }
      }
    }
  }
}

// ---------------------------------------------------------------- dense WMMA linear: [M,K] f16 x [K,NCOL] + bias
// hlin has sentinel zero row M; B fragments zero-padded to NT*16 = 64 cols; one col tile per wave.
template <int K, int NCOL>
__global__ __launch_bounds__(128)
void k_linear_wmma(const f16* __restrict__ hlin, const unsigned* __restrict__ Bp,
                   const float* __restrict__ bias, float* __restrict__ out, int M) {
  constexpr int KT = K / 32;
  constexpr int NT = 4;

  __shared__ __align__(16) unsigned Atile[2][16 * ATILE_STRIDE];

  const int t    = threadIdx.x;
  const int wave = __builtin_amdgcn_readfirstlane(t >> 5);
  const int lane = t & 31;
  const int m    = lane & 15;
  const int hh   = lane >> 4;
  const int tile = blockIdx.x;

  const v8f vz = {0.f, 0.f, 0.f, 0.f, 0.f, 0.f, 0.f, 0.f};
  v8f acc = vz;

  const int rstage = t >> 3;
  const int dpair  = t & 7;
  const int rg0    = tile * 16 + rstage;
  const int rg     = (rg0 < M) ? rg0 : M;               // sentinel row -> zeros

  unsigned ldsA[2];
  ldsA[0] = (unsigned)(uintptr_t)(void*)&Atile[0][rstage * ATILE_STRIDE + dpair * 2];
  ldsA[1] = (unsigned)(uintptr_t)(void*)&Atile[1][rstage * ATILE_STRIDE + dpair * 2];

  auto stage = [&](int kt) {
    const f16* gp = hlin + (unsigned)rg * K + kt * 32u + dpair * 4u;
    asm volatile("global_load_async_to_lds_b64 %0, %1, off"
                 :: "v"(ldsA[kt & 1]), "v"((unsigned long long)(uintptr_t)gp)
                 : "memory");
  };

  stage(0);
  for (int kt = 0; kt < KT; ++kt) {
    asm volatile("s_wait_asynccnt 0x0" ::: "memory");
    __syncthreads();
    if (kt + 1 < KT) {
      stage(kt + 1);
      __builtin_prefetch(Bp + (((size_t)(kt + 1) * NT + wave) * 32 + lane) * 8, 0, 1);
    }
    const unsigned* cur = &Atile[kt & 1][0];
    union { v16h h; uint4 q[2]; } af;
    af.q[0] = ((const uint4*)cur)[m * (ATILE_STRIDE / 4) + 2 * hh];
    af.q[1] = ((const uint4*)cur)[m * (ATILE_STRIDE / 4) + 2 * hh + 1];
    const uint4* bp = (const uint4*)(Bp + (((size_t)kt * NT + wave) * 32 + lane) * 8);
    union { v16h h; uint4 q[2]; } bf_;
    bf_.q[0] = bp[0];
    bf_.q[1] = bp[1];
    acc = __builtin_amdgcn_wmma_f32_16x16x32_f16(
        false, af.h, false, bf_.h, (short)0, acc, false, false);
  }

  const int col = wave * 16 + m;
  if (col < NCOL) {                                     // lane-divergent: stores only
    float bv = bias[col];
    #pragma unroll
    for (int vv = 0; vv < 8; ++vv) {
      int row = tile * 16 + vv + 8 * hh;
      if (row < M) out[(size_t)row * NCOL + col] = acc[vv] + bv;
    }
  }
}

// ---------------------------------------------------------------- concat helpers
__global__ void k_copyf(const float* __restrict__ f, float* __restrict__ cat,
                        int N, int C, int D) {
  size_t total = (size_t)N * C;
  size_t stride = (size_t)gridDim.x * blockDim.x;
  for (size_t i = (size_t)blockIdx.x * blockDim.x + threadIdx.x; i < total; i += stride) {
    int c = (int)(i % (size_t)C);
    int n = (int)(i / (size_t)C);
    cat[(size_t)n * D + c] = f[i];
  }
}

__global__ void k_unpool(const float* __restrict__ deep, const int* __restrict__ up,
                         float* __restrict__ cat, int N, int Ndeep, int Cdeep,
                         int D, int coff) {
  size_t total = (size_t)N * Cdeep;
  size_t stride = (size_t)gridDim.x * blockDim.x;
  for (size_t i = (size_t)blockIdx.x * blockDim.x + threadIdx.x; i < total; i += stride) {
    int c = (int)(i % (size_t)Cdeep);
    int n = (int)(i / (size_t)Cdeep);
    float s = 0.f;
    #pragma unroll
    for (int j = 0; j < 8; ++j) {
      int idx = up[n * 8 + j];
      if (idx < Ndeep) s += deep[(size_t)idx * Cdeep + c];
    }
    cat[(size_t)n * D + coff + c] = s;
  }
}

// ---------------------------------------------------------------- host orchestration
extern "C" void kernel_launch(void* const* d_in, const int* in_sizes, int n_in,
                              void* d_out, int out_size, void* d_ws, size_t ws_size,
                              hipStream_t stream) {
  (void)out_size; (void)ws_size;
  static const int CH[5] = {32, 64, 96, 128, 160};
  static const int DCH[5] = {480, 448, 384, 288, 160};

  const float *x = nullptr, *w_in = nullptr;
  const float *g[5][2], *b[5][2], *w[5][2];
  const float *gd[4], *bd[4], *wd[4];
  const float *gf = nullptr, *bfp = nullptr, *w_lin = nullptr, *b_lin = nullptr;
  const int *nb[5], *down[4], *up[4];
  int downIdx[4]; int xIdx;

  auto F = [&](int i) { return (const float*)d_in[i]; };
  auto I = [&](int i) { return (const int*)d_in[i]; };

  // Detect flattening order: insertion order puts x (100000 elems) first;
  // jax-tree sorted order puts it last (idx['down0'] first, size N1*27).
  bool sortedOrder = !(n_in > 0 && in_sizes[0] == 100000);
  if (sortedOrder) {
    for (int l = 0; l < 4; ++l) { down[l] = I(l); downIdx[l] = l; }
    for (int l = 0; l < 5; ++l) nb[l] = I(4 + l);
    for (int l = 0; l < 4; ++l) up[l] = I(9 + l);
    int p = 13;
    for (int l = 0; l < 5; ++l) { b[l][0] = F(p++); b[l][1] = F(p++); }
    b_lin = F(p++);
    for (int l = 0; l < 4; ++l) bd[l] = F(p++);
    bfp = F(p++);
    for (int l = 0; l < 5; ++l) { g[l][0] = F(p++); g[l][1] = F(p++); }
    for (int l = 0; l < 4; ++l) gd[l] = F(p++);
    gf = F(p++);
    for (int l = 0; l < 5; ++l) { w[l][0] = F(p++); w[l][1] = F(p++); }
    w_in = F(p++); w_lin = F(p++);
    for (int l = 0; l < 4; ++l) wd[l] = F(p++);
    xIdx = p; x = F(p++);
  } else {
    xIdx = 0; x = F(0); w_in = F(1);
    int p = 2;
    for (int l = 0; l < 5; ++l) {
      g[l][0] = F(p++); b[l][0] = F(p++); w[l][0] = F(p++);
      g[l][1] = F(p++); b[l][1] = F(p++); w[l][1] = F(p++);
      if (l < 4) { gd[l] = F(p++); bd[l] = F(p++); wd[l] = F(p++); }
    }
    gf = F(p++); bfp = F(p++); w_lin = F(p++); b_lin = F(p++);
    for (int l = 0; l < 4; ++l) {
      nb[l] = I(p); down[l] = I(p + 1); downIdx[l] = p + 1; up[l] = I(p + 2); p += 3;
    }
    nb[4] = I(p++);
  }

  int N[5];
  N[0] = in_sizes[xIdx];
  for (int l = 0; l < 4; ++l) N[l + 1] = in_sizes[downIdx[l]] / 27;

  // bump-allocate workspace
  char* wsp = (char*)d_ws;
  size_t off = 0;
  auto alloc = [&](size_t bytes) -> void* {
    void* r = wsp + off;
    off = (off + bytes + 255) & ~(size_t)255;
    return r;
  };
  float* f[5]; f16* h[5]; float* cat[5];
  for (int l = 0; l < 5; ++l) {
    f[l] = (float*)alloc((size_t)N[l] * CH[l] * sizeof(float));
    h[l] = (f16*)alloc((size_t)(N[l] + 1) * CH[l] * sizeof(f16));
  }
  for (int l = 0; l < 4; ++l) cat[l] = (float*)alloc((size_t)N[l] * DCH[l] * sizeof(float));
  cat[4] = f[4];
  f16* hlin = (f16*)alloc((size_t)(N[0] + 1) * 480 * sizeof(f16));
  float* stats = (float*)alloc(2 * 480 * sizeof(float));
  unsigned* Bp = (unsigned*)alloc((size_t)27 * 160 * 160 * 2);  // max packed-B bytes

  auto cdiv = [](size_t a, size_t bq) { return (int)((a + bq - 1) / bq); };

  auto bnrelu = [&](const float* src, const float* gg, const float* bb,
                    f16* dst, int Nn, int C) {
    k_zero<<<cdiv(2 * C, 256), 256, 0, stream>>>(stats, 2 * C);
    k_bn_stats<<<cdiv(Nn, 512), 256, 0, stream>>>(src, stats, Nn, C, 512);
    k_bn_relu_cast<<<cdiv((size_t)(Nn + 1) * C, 256), 256, 0, stream>>>(src, stats, gg, bb, dst, Nn, C);
  };

  auto gather = [&](const f16* hsrc, const int* nbp, const float* wsrc, float* outp,
                    int M, int Nsrc, int Cin, int Cout) {
    int KT = 27 * Cin / 32, NT = Cout / 16;
    k_pack_w<<<cdiv((size_t)KT * NT * 32, 256), 256, 0, stream>>>(wsrc, Bp, 27 * Cin, Cout, NT);
    int blocks = cdiv(M, 16);
    if (false) {}
#define GCASE(ci, co) \
    else if (Cin == ci && Cout == co) \
      k_gather_wmma<ci, co><<<blocks, 128, 0, stream>>>(hsrc, nbp, Bp, outp, M, Nsrc)
    GCASE(32, 32); GCASE(64, 64); GCASE(96, 96); GCASE(128, 128); GCASE(160, 160);
    GCASE(32, 64); GCASE(64, 96); GCASE(96, 128); GCASE(128, 160);
#undef GCASE
  };

  // ---- forward ----
  k_conv_in<<<cdiv((size_t)N[0] * 32, 256), 256, 0, stream>>>(x, nb[0], w_in, f[0], N[0]);

  for (int l = 0; l < 5; ++l) {
    for (int r = 0; r < 2; ++r) {
      bnrelu(f[l], g[l][r], b[l][r], h[l], N[l], CH[l]);
      gather(h[l], nb[l], w[l][r], f[l], N[l], N[l], CH[l], CH[l]);
    }
    if (l < 4) {
      bnrelu(f[l], gd[l], bd[l], h[l], N[l], CH[l]);
      gather(h[l], down[l], wd[l], f[l + 1], N[l + 1], N[l], CH[l], CH[l + 1]);
    }
  }

  // ---- decoder: concat(f_l, unpool(deep)) from deep to shallow ----
  for (int l = 3; l >= 0; --l) {
    k_copyf<<<cdiv((size_t)N[l] * CH[l], 256), 256, 0, stream>>>(f[l], cat[l], N[l], CH[l], DCH[l]);
    k_unpool<<<cdiv((size_t)N[l] * DCH[l + 1], 256), 256, 0, stream>>>(
        cat[l + 1], up[l], cat[l], N[l], N[l + 1], DCH[l + 1], DCH[l], CH[l]);
  }

  // ---- final BN+ReLU (f16) + WMMA linear ----
  bnrelu(cat[0], gf, bfp, hlin, N[0], 480);
  k_pack_w<<<cdiv((size_t)(480 / 32) * 4 * 32, 256), 256, 0, stream>>>(w_lin, Bp, 480, 50, 4);
  k_linear_wmma<480, 50><<<cdiv(N[0], 16), 128, 0, stream>>>(hlin, Bp, b_lin, (float*)d_out, N[0]);
}
